// genconv_morph_9732395893314
// MI455X (gfx1250) — matrix operand order repaired and checked by reading, not compiled, not run on hardware
//
#include <hip/hip_runtime.h>

#define NN    50000
#define EE    400000
#define FIN   514
#define HH    128
#define HH2   256
#define NL    3
#define NCLS  4

typedef __attribute__((ext_vector_type(16))) _Float16 v16h;
typedef __attribute__((ext_vector_type(8)))  _Float16 v8h;
typedef __attribute__((ext_vector_type(8)))  float    v8f;
typedef __attribute__((ext_vector_type(4)))  unsigned int v4u;
typedef __attribute__((ext_vector_type(8)))  int      v8i;
typedef __attribute__((ext_vector_type(4)))  int      v4i;

#define A_STRIDE 40   // 32 + 8 f16 pad, keeps 16B alignment for kbase in {0,8}

__device__ __forceinline__ v8f wmma_f16(v16h a, v16h b, v8f c) {
    return __builtin_amdgcn_wmma_f32_16x16x32_f16(
        false, a, false, b, (short)0, c, false, false);
}

// A fragment per ISA "16-bit A-Matrix 16x32" layout
__device__ __forceinline__ v16h load_a_frag(const _Float16* As, int lane) {
    int m  = lane & 15;
    int kb = (lane < 16) ? 0 : 8;
    union { v16h v; v8h h[2]; } f;
    f.h[0] = *(const v8h*)(As + m * A_STRIDE + kb);
    f.h[1] = *(const v8h*)(As + m * A_STRIDE + kb + 16);
    return f.v;
}

// packed B: packed[(ntile*32 + lane)*16 + j] = B[(lane<16?0:8) + (j<8 ? j : j+8)][ntile*16 + lane%16]
__device__ __forceinline__ v16h load_b_frag(const _Float16* Bp, int ntile, int lane) {
    return *(const v16h*)(Bp + (ntile * 32 + lane) * 16);
}
__device__ __forceinline__ int b_pack_idx(int kk, int n) {
    return (((n >> 4) * 32) + ((kk & 8) << 1) + (n & 15)) * 16 + (kk & 7) + ((kk & 16) >> 1);
}

__device__ __forceinline__ void atomicMaxF(float* addr, float val) {
    if (val >= 0.f) atomicMax((int*)addr, __float_as_int(val));
    else            atomicMin((unsigned int*)addr, __float_as_uint(val));
}

// -------- CDNA5 TDM: issue a 2D f16 tile load (16 rows x 32 cols, row stride `srcStride`)
// into LDS at `ldsoff`, padded to A_STRIDE rows via D# pad fields.
__device__ __forceinline__ void tdm_load_a_tile(unsigned ldsoff, const _Float16* gsrc,
                                                unsigned srcStride, unsigned nrowsTot) {
    unsigned long long ga = (unsigned long long)(uintptr_t)gsrc;
    v4u g0;
    g0.x = 1u;                                  // count=1, user descriptor
    g0.y = ldsoff;                              // lds_addr
    g0.z = (unsigned)ga;                        // global_addr[31:0]
    g0.w = (unsigned)((ga >> 32) & 0x1FFFFFFu) | (2u << 30);  // addr[56:32] | type=2
    v8i g1;
    // data_size=1(2B) | pad_enable | pad_interval=3 (16 DW = 64B row) | pad_amount=3 (4 DW = 16B)
    g1[0] = (int)(0x10000u | (1u << 20) | (3u << 22) | (3u << 25));
    g1[1] = (int)((srcStride & 0xFFFFu) << 16);              // tensor_dim0[15:0]
    g1[2] = (int)(((srcStride >> 16) & 0xFFFFu) | ((nrowsTot & 0xFFFFu) << 16));
    g1[3] = (int)(((nrowsTot >> 16) & 0xFFFFu) | (32u << 16)); // tile_dim0=32
    g1[4] = 16;                                              // tile_dim1=16
    g1[5] = (int)srcStride;                                  // tensor_dim0_stride[31:0]
    g1[6] = 0;
    g1[7] = 0;
    v4i z4 = {0, 0, 0, 0};
    v8i z8 = {0, 0, 0, 0, 0, 0, 0, 0};
    __builtin_amdgcn_tensor_load_to_lds(g0, g1, z4, z4, z8, 0);
}

// -------- CDNA5 async global->LDS 16B copy + wait (ASYNCcnt)
__device__ __forceinline__ void async_copy_b128(unsigned ldsoff, const void* gptr) {
    asm volatile("global_load_async_to_lds_b128 %0, %1, off"
                 :: "v"(ldsoff), "v"(gptr) : "memory");
}
__device__ __forceinline__ void wait_asynccnt0() {
    asm volatile("s_wait_asynccnt 0x0" ::: "memory");
}

// ---------------------------------------------------------------- weight cvt
__global__ __launch_bounds__(256) void k_cvt(const float* __restrict__ s,
                                             _Float16* __restrict__ d, int n) {
    int i = blockIdx.x * 256 + threadIdx.x;
    if (i < n) d[i] = (_Float16)s[i];
}

// ------------------------------------------------- node encoder: [N,514]@[514,128]+bn
__global__ __launch_bounds__(128) void k_gemm_enc(const float* __restrict__ nodex,
                                                  const _Float16* __restrict__ Wn16,
                                                  const float* __restrict__ bn,
                                                  float* __restrict__ X0) {
    __shared__ __align__(32) _Float16 As[16 * A_STRIDE];
    __shared__ __align__(32) _Float16 Bp[8 * 32 * 16];
    int m0 = blockIdx.x * 16;
    int tid = threadIdx.x, lane = tid & 31, wave = tid >> 5;
    v8f acc0 = {}, acc1 = {};
    // 16 full K-tiles, no bounds checks (512 = 16*32)
    for (int k0 = 0; k0 < 512; k0 += 32) {
        __syncthreads();
        for (int idx = tid; idx < 256; idx += 128) {   // float2 loads, 8B aligned
            int m = idx >> 4, kk = (idx & 15) * 2;
            float2 v = *(const float2*)(nodex + (size_t)(m0 + m) * FIN + k0 + kk);
            As[m * A_STRIDE + kk]     = (_Float16)v.x;
            As[m * A_STRIDE + kk + 1] = (_Float16)v.y;
        }
        for (int idx = tid; idx < 32 * HH; idx += 128) {
            int kk = idx >> 7, n = idx & 127;
            Bp[b_pack_idx(kk, n)] = Wn16[(k0 + kk) * HH + n];
        }
        __builtin_prefetch(&Wn16[(k0 + 32) * HH], 0, 1);
        __syncthreads();
        v16h a = load_a_frag(As, lane);
        acc0 = wmma_f16(a, load_b_frag(Bp, 2 * wave + 0, lane), acc0);
        acc1 = wmma_f16(a, load_b_frag(Bp, 2 * wave + 1, lane), acc1);
    }
    // tail tile: K = 512..513 (2 valid columns, rest zero)
    __syncthreads();
    for (int idx = tid; idx < 512; idx += 128) {
        int m = idx >> 5, kk = idx & 31;
        As[m * A_STRIDE + kk] = (kk < FIN - 512)
            ? (_Float16)nodex[(size_t)(m0 + m) * FIN + 512 + kk] : (_Float16)0.f;
    }
    for (int idx = tid; idx < 32 * HH; idx += 128) {
        int kk = idx >> 7, n = idx & 127;
        Bp[b_pack_idx(kk, n)] = (kk < FIN - 512) ? Wn16[(512 + kk) * HH + n] : (_Float16)0.f;
    }
    __syncthreads();
    {
        v16h a = load_a_frag(As, lane);
        acc0 = wmma_f16(a, load_b_frag(Bp, 2 * wave + 0, lane), acc0);
        acc1 = wmma_f16(a, load_b_frag(Bp, 2 * wave + 1, lane), acc1);
    }
    int n_ = lane & 15, mb = (lane < 16) ? 0 : 8;
    for (int i = 0; i < 8; i++) {
        int m = m0 + mb + i;
        int c0 = (2 * wave + 0) * 16 + n_;
        int c1 = (2 * wave + 1) * 16 + n_;
        X0[(size_t)m * HH + c0] = acc0[i] + bn[c0];
        X0[(size_t)m * HH + c1] = acc1[i] + bn[c1];
    }
}

// ----------------------------------- per-layer prologue
__global__ __launch_bounds__(128) void k_pre(const float* __restrict__ X0,
                                             const float* __restrict__ g,
                                             const float* __restrict__ b,
                                             float* __restrict__ Y, float* __restrict__ SMAX,
                                             float* __restrict__ DEN, float* __restrict__ AGG,
                                             int do_ln) {
    __shared__ float red[8];
    int row = blockIdx.x, t = threadIdx.x, lane = t & 31, wave = t >> 5;
    float v = X0[(size_t)row * HH + t];
    if (do_ln) {
        float s = v, s2 = v * v;
        for (int o = 16; o > 0; o >>= 1) { s += __shfl_xor(s, o, 32); s2 += __shfl_xor(s2, o, 32); }
        if (lane == 0) { red[wave] = s; red[4 + wave] = s2; }
        __syncthreads();
        float S  = red[0] + red[1] + red[2] + red[3];
        float S2 = red[4] + red[5] + red[6] + red[7];
        float mu = S * (1.f / HH), var = S2 * (1.f / HH) - mu * mu;
        v = fmaxf((v - mu) * rsqrtf(var + 1e-5f) * g[t] + b[t], 0.f);
    }
    size_t i = (size_t)row * HH + t;
    Y[i] = v;
    SMAX[i] = -3.0e38f;
    DEN[i] = 0.f;
    AGG[i] = 0.f;
}

// ---------------------------------------------------------------- edge pass 1: segment max
__global__ __launch_bounds__(256) void k_edge1(const float* __restrict__ Y,
                                               const float* __restrict__ ear,
                                               const int* __restrict__ src,
                                               const int* __restrict__ dst,
                                               const float* __restrict__ We,
                                               const float* __restrict__ be,
                                               const float* __restrict__ tptr,
                                               float* __restrict__ SMAX) {
    int e = blockIdx.x * 2 + (threadIdx.x >> 7);
    int f = threadIdx.x & 127;
    int s = src[e], d = dst[e];
    float tt = *tptr;
    float a0 = ear[(size_t)e * 2 + 0], a1 = ear[(size_t)e * 2 + 1];
    float ea = a0 * We[f] + a1 * We[HH + f] + be[f];
    float msg = fmaxf(Y[(size_t)s * HH + f] + ea, 0.f) + 1e-7f;
    atomicMaxF(&SMAX[(size_t)d * HH + f], msg * tt);
}

// ------------------------------------------- edge pass 2
__global__ __launch_bounds__(256) void k_edge2(const float* __restrict__ Y,
                                               const float* __restrict__ ear,
                                               const int* __restrict__ src,
                                               const int* __restrict__ dst,
                                               const float* __restrict__ We,
                                               const float* __restrict__ be,
                                               const float* __restrict__ tptr,
                                               const float* __restrict__ SMAX,
                                               float* __restrict__ DEN, float* __restrict__ AGG) {
    int e = blockIdx.x * 2 + (threadIdx.x >> 7);
    int f = threadIdx.x & 127;
    int s = src[e], d = dst[e];
    float tt = *tptr;
    float a0 = ear[(size_t)e * 2 + 0], a1 = ear[(size_t)e * 2 + 1];
    float ea = a0 * We[f] + a1 * We[HH + f] + be[f];
    float msg = fmaxf(Y[(size_t)s * HH + f] + ea, 0.f) + 1e-7f;
    float sc = msg * tt;
    float w = __expf(sc - SMAX[(size_t)d * HH + f]);
    atomicAdd(&DEN[(size_t)d * HH + f], w);
    atomicAdd(&AGG[(size_t)d * HH + f], msg * w);
}

// ---------------------------------- combine
__global__ __launch_bounds__(256) void k_combine(const float* __restrict__ AGG,
                                                 const float* __restrict__ DEN,
                                                 const float* __restrict__ Y,
                                                 _Float16* __restrict__ OUT16) {
    size_t i = (size_t)blockIdx.x * 256 + threadIdx.x;
    OUT16[i] = (_Float16)(AGG[i] / (DEN[i] + 1e-16f) + Y[i]);
}

// -------------------- MLP1 (A-tile via TDM): [N,128]@[128,256]+b1 -> LN -> relu -> H16
__global__ __launch_bounds__(256) void k_mlp1(const _Float16* __restrict__ OUT16,
                                              const _Float16* __restrict__ W116,
                                              const float* __restrict__ b1,
                                              const float* __restrict__ g1,
                                              const float* __restrict__ bt1,
                                              _Float16* __restrict__ H16) {
    __shared__ __align__(32) _Float16 As[16 * A_STRIDE];
    __shared__ __align__(32) _Float16 Bp[16 * 32 * 16];
    __shared__ float Cb[16 * HH2];
    int m0 = blockIdx.x * 16;
    int tid = threadIdx.x, lane = tid & 31, wave = tid >> 5;
    v8f acc0 = {}, acc1 = {};
    unsigned asOff = (unsigned)(uintptr_t)&As[0];
    for (int k0 = 0; k0 < HH; k0 += 32) {
        __syncthreads();
        if (wave == 0) {  // TDM DMA: 16x32 f16 tile, LDS-padded to A_STRIDE
            tdm_load_a_tile(asOff, OUT16 + (size_t)m0 * HH + k0, HH, NN);
        }
        for (int idx = tid; idx < 32 * HH2; idx += 256) {
            int kk = idx >> 8, n = idx & 255;
            Bp[b_pack_idx(kk, n)] = W116[(k0 + kk) * HH2 + n];
        }
        if (k0 + 32 < HH) __builtin_prefetch(&W116[(k0 + 32) * HH2], 0, 1);
        if (wave == 0) __builtin_amdgcn_s_wait_tensorcnt(0);
        __syncthreads();
        v16h a = load_a_frag(As, lane);
        acc0 = wmma_f16(a, load_b_frag(Bp, 2 * wave + 0, lane), acc0);
        acc1 = wmma_f16(a, load_b_frag(Bp, 2 * wave + 1, lane), acc1);
    }
    int n_ = lane & 15, mb = (lane < 16) ? 0 : 8;
    for (int i = 0; i < 8; i++) {
        int c0 = (2 * wave + 0) * 16 + n_;
        int c1 = (2 * wave + 1) * 16 + n_;
        Cb[(mb + i) * HH2 + c0] = acc0[i] + b1[c0];
        Cb[(mb + i) * HH2 + c1] = acc1[i] + b1[c1];
    }
    __syncthreads();
    for (int r = wave; r < 16; r += 8) {
        float s = 0.f, s2 = 0.f;
        for (int j = lane; j < HH2; j += 32) { float v = Cb[r * HH2 + j]; s += v; s2 += v * v; }
        for (int o = 16; o > 0; o >>= 1) { s += __shfl_xor(s, o, 32); s2 += __shfl_xor(s2, o, 32); }
        float mu = s * (1.f / HH2), var = s2 * (1.f / HH2) - mu * mu;
        float inv = rsqrtf(var + 1e-5f);
        for (int j = lane; j < HH2; j += 32) {
            float v = (Cb[r * HH2 + j] - mu) * inv * g1[j] + bt1[j];
            H16[(size_t)(m0 + r) * HH2 + j] = (_Float16)fmaxf(v, 0.f);
        }
    }
}

// -------------------- MLP2 (A-tile via async LDS copies): [N,256]@[256,128]+b2 (+res)
__global__ __launch_bounds__(128) void k_mlp2(const _Float16* __restrict__ H16,
                                              const _Float16* __restrict__ W216,
                                              const float* __restrict__ b2,
                                              float* __restrict__ X0, int residual) {
    __shared__ __align__(32) _Float16 As[16 * A_STRIDE];
    __shared__ __align__(32) _Float16 Bp[8 * 32 * 16];
    int m0 = blockIdx.x * 16;
    int tid = threadIdx.x, lane = tid & 31, wave = tid >> 5;
    v8f acc0 = {}, acc1 = {};
    for (int k0 = 0; k0 < HH2; k0 += 32) {
        __syncthreads();
        if (tid < 64) {   // async global->LDS: 64 lanes x 16B = 16x32 f16 tile
            int m = tid >> 2, part = tid & 3;
            unsigned ldsoff = (unsigned)(uintptr_t)(As + m * A_STRIDE + part * 8);
            const _Float16* g = H16 + (size_t)(m0 + m) * HH2 + k0 + part * 8;
            async_copy_b128(ldsoff, g);
        }
        for (int idx = tid; idx < 32 * HH; idx += 128) {
            int kk = idx >> 7, n = idx & 127;
            Bp[b_pack_idx(kk, n)] = W216[(k0 + kk) * HH + n];
        }
        if (k0 + 32 < HH2) __builtin_prefetch(&W216[(k0 + 32) * HH], 0, 1);
        wait_asynccnt0();
        __syncthreads();
        v16h a = load_a_frag(As, lane);
        acc0 = wmma_f16(a, load_b_frag(Bp, 2 * wave + 0, lane), acc0);
        acc1 = wmma_f16(a, load_b_frag(Bp, 2 * wave + 1, lane), acc1);
    }
    int n_ = lane & 15, mb = (lane < 16) ? 0 : 8;
    for (int i = 0; i < 8; i++) {
        int m = m0 + mb + i;
        int c0 = (2 * wave + 0) * 16 + n_;
        int c1 = (2 * wave + 1) * 16 + n_;
        float r0 = residual ? X0[(size_t)m * HH + c0] : 0.f;
        float r1 = residual ? X0[(size_t)m * HH + c1] : 0.f;
        X0[(size_t)m * HH + c0] = r0 + acc0[i] + b2[c0];
        X0[(size_t)m * HH + c1] = r1 + acc1[i] + b2[c1];
    }
}

// ------------- final head
__global__ __launch_bounds__(128) void k_final(const float* __restrict__ X0,
                                               const float* __restrict__ g,
                                               const float* __restrict__ b,
                                               const float* __restrict__ Wout,
                                               const float* __restrict__ bout,
                                               float* __restrict__ out) {
    __shared__ float red[8];
    __shared__ float cls[16];
    __shared__ float logits[4];
    int row = blockIdx.x, t = threadIdx.x, lane = t & 31, wave = t >> 5;
    float v = X0[(size_t)row * HH + t];
    float s = v, s2 = v * v;
    for (int o = 16; o > 0; o >>= 1) { s += __shfl_xor(s, o, 32); s2 += __shfl_xor(s2, o, 32); }
    if (lane == 0) { red[wave] = s; red[4 + wave] = s2; }
    __syncthreads();
    float S  = red[0] + red[1] + red[2] + red[3];
    float S2 = red[4] + red[5] + red[6] + red[7];
    float mu = S * (1.f / HH), var = S2 * (1.f / HH) - mu * mu;
    v = fmaxf((v - mu) * rsqrtf(var + 1e-5f) * g[t] + b[t], 0.f);
    float c0 = v * Wout[t * 4 + 0], c1 = v * Wout[t * 4 + 1];
    float c2 = v * Wout[t * 4 + 2], c3 = v * Wout[t * 4 + 3];
    for (int o = 16; o > 0; o >>= 1) {
        c0 += __shfl_xor(c0, o, 32); c1 += __shfl_xor(c1, o, 32);
        c2 += __shfl_xor(c2, o, 32); c3 += __shfl_xor(c3, o, 32);
    }
    if (lane == 0) { cls[wave * 4 + 0] = c0; cls[wave * 4 + 1] = c1;
                     cls[wave * 4 + 2] = c2; cls[wave * 4 + 3] = c3; }
    __syncthreads();
    if (t < 4) logits[t] = cls[t] + cls[4 + t] + cls[8 + t] + cls[12 + t] + bout[t];
    __syncthreads();
    if (t < 4) {
        float m = fmaxf(fmaxf(logits[0], logits[1]), fmaxf(logits[2], logits[3]));
        float sum = __expf(logits[0] - m) + __expf(logits[1] - m) +
                    __expf(logits[2] - m) + __expf(logits[3] - m);
        out[(size_t)row * 4 + t] = logits[t] - m - __logf(sum);
    }
}

extern "C" void kernel_launch(void* const* d_in, const int* in_sizes, int n_in,
                              void* d_out, int out_size, void* d_ws, size_t ws_size,
                              hipStream_t stream) {
    const float* node_x = (const float*)d_in[0];
    const float* ear    = (const float*)d_in[1];
    const int*   ei     = (const int*)d_in[2];
    const float* Wn     = (const float*)d_in[3];
    const float* bn     = (const float*)d_in[4];
    const float* We     = (const float*)d_in[5];
    const float* be     = (const float*)d_in[6];
    const float* tptr   = (const float*)d_in[7];
    const float* W1     = (const float*)d_in[8];
    const float* b1     = (const float*)d_in[9];
    const float* lng    = (const float*)d_in[10];
    const float* lnb    = (const float*)d_in[11];
    const float* W2     = (const float*)d_in[12];
    const float* b2     = (const float*)d_in[13];
    const float* dng    = (const float*)d_in[14];
    const float* dnb    = (const float*)d_in[15];
    const float* Wout   = (const float*)d_in[16];
    const float* bout   = (const float*)d_in[17];
    float* out = (float*)d_out;

    char* ws = (char*)d_ws;
    size_t off = 0;
    auto alloc = [&](size_t bytes) -> void* {
        void* p = (void*)(ws + off);
        off = (off + bytes + 255) & ~(size_t)255;
        return p;
    };
    float*     X0    = (float*)alloc((size_t)NN * HH * 4);
    float*     Y     = (float*)alloc((size_t)NN * HH * 4);
    float*     SMAX  = (float*)alloc((size_t)NN * HH * 4);
    float*     DEN   = (float*)alloc((size_t)NN * HH * 4);
    float*     AGG   = (float*)alloc((size_t)NN * HH * 4);
    _Float16*  OUT16 = (_Float16*)alloc((size_t)NN * HH * 2);
    _Float16*  H16   = (_Float16*)alloc((size_t)NN * HH2 * 2);
    _Float16*  Wn16  = (_Float16*)alloc((size_t)FIN * HH * 2);
    _Float16*  W116  = (_Float16*)alloc((size_t)NL * HH * HH2 * 2);
    _Float16*  W216  = (_Float16*)alloc((size_t)NL * HH2 * HH * 2);

    int nWn = FIN * HH, nW1 = NL * HH * HH2, nW2 = NL * HH2 * HH;
    k_cvt<<<(nWn + 255) / 256, 256, 0, stream>>>(Wn, Wn16, nWn);
    k_cvt<<<(nW1 + 255) / 256, 256, 0, stream>>>(W1, W116, nW1);
    k_cvt<<<(nW2 + 255) / 256, 256, 0, stream>>>(W2, W216, nW2);

    const int MT = NN / 16;  // 3125, exact
    k_gemm_enc<<<MT, 128, 0, stream>>>(node_x, Wn16, bn, X0);

    for (int l = 0; l < NL; l++) {
        k_pre<<<NN, 128, 0, stream>>>(X0, dng + l * HH, dnb + l * HH,
                                      Y, SMAX, DEN, AGG, (l > 0) ? 1 : 0);
        k_edge1<<<EE / 2, 256, 0, stream>>>(Y, ear, ei, ei + EE, We, be, tptr + l, SMAX);
        k_edge2<<<EE / 2, 256, 0, stream>>>(Y, ear, ei, ei + EE, We, be, tptr + l,
                                            SMAX, DEN, AGG);
        k_combine<<<(NN * HH) / 256, 256, 0, stream>>>(AGG, DEN, Y, OUT16);
        k_mlp1<<<MT, 256, 0, stream>>>(OUT16, W116 + (size_t)l * HH * HH2,
                                       b1 + l * HH2, lng + l * HH2, lnb + l * HH2, H16);
        k_mlp2<<<MT, 128, 0, stream>>>(H16, W216 + (size_t)l * HH2 * HH,
                                       b2 + l * HH, X0, (l > 0) ? 1 : 0);
    }
    k_final<<<NN, 128, 0, stream>>>(X0, dng, dnb, Wout, bout, out);
}
